// CGNe3_79448305042186
// MI455X (gfx1250) — compile-verified, matrix operand order checked by reading, not run on hardware
//
#include <hip/hip_runtime.h>
#include <hip/hip_bf16.h>
#include <math.h>

#define N_NODES 50000
#define N_EDGES 400000
#define H 128
#define FN 92
#define FE 41
#define LYR 3
#define PC 8
#define PD 16
#define SC 8
#define SD 32
#define BGR 64
#define EPSV 1e-5f

typedef _Float16 v16h __attribute__((ext_vector_type(16)));
typedef _Float16 v8h  __attribute__((ext_vector_type(8)));
typedef float    v8f  __attribute__((ext_vector_type(8)));

__device__ __forceinline__ float rcp_f(float x){ return __builtin_amdgcn_rcpf(x); }
__device__ __forceinline__ float sigm_f(float x){ return rcp_f(1.0f + __expf(-x)); }
__device__ __forceinline__ float silu_f(float x){ return x * sigm_f(x); }

// ---------------------------------------------------------------------------
// Generic WMMA GEMM: out[M,128] = EPI(A[M,K] @ W[K,128])
// Block = 256 threads (8 waves), tile = 32 rows x 128 cols.
// Wave (wr,wc) computes 16x32 via two v_wmma_f32_16x16x32_f16 accumulators.
// LDS: A row-major [r][k]; W transposed [n][k] so every lane's fragment is two
// contiguous 16-byte runs -> ds_load_b128 instead of scalar u16 loads.
// EPI: 0 = raw, 1 = +bias, 2 = silu(+bias), 3 = silu((+bias)*gamma+beta)
// ---------------------------------------------------------------------------
template<int EPI>
__global__ __launch_bounds__(256)
void gemm128_wmma(const float* __restrict__ A, int lda, int K,
                  const float* __restrict__ W,
                  const float* __restrict__ bias,
                  const float* __restrict__ gamma,
                  const float* __restrict__ beta,
                  float* __restrict__ out, int M)
{
    __shared__ __align__(16) _Float16 sBt[128*128];  // W^T tile: [n][k], 32 KB
    __shared__ __align__(16) _Float16 sA [32*128];   // A tile:   [r][k], 8 KB
    const int tid = threadIdx.x;
    const int blockRow = blockIdx.x * 32;

    for (int idx = tid; idx < 128*128; idx += 256){
        int k = idx >> 7, c = idx & 127;             // coalesced read of W[k][c]
        sBt[c*128 + k] = (k < K) ? (_Float16)W[k*128 + c] : (_Float16)0.0f;
    }
    for (int idx = tid; idx < 32*128; idx += 256){
        int r = idx >> 7, k = idx & 127;
        int gr = blockRow + r;
        sA[idx] = (gr < M && k < K) ? (_Float16)A[(long)gr*lda + k] : (_Float16)0.0f;
    }
    __syncthreads();

    const int lane = tid & 31;
    const int wid  = tid >> 5;
    const int wr = wid >> 2, wc = wid & 3;
    const int hi  = (lane >> 4) & 1;     // half-wave K-striping select
    const int l15 = lane & 15;
    const int Kp  = (K + 31) & ~31;

    // per-lane fragment bases: elements e=0..7 at K = kk+8*hi+e,
    //                          elements e=8..15 at K = kk+16+8*hi+(e-8)
    const int aBase  = (wr*16 + l15)*128 + (hi << 3);
    const int b0Base = (wc*32 + l15)*128 + (hi << 3);
    const int b1Base = (wc*32 + 16 + l15)*128 + (hi << 3);

    v8f acc0 = {}, acc1 = {};
    for (int kk = 0; kk < Kp; kk += 32){
        v8h aLo = *(const v8h*)&sA [aBase  + kk];
        v8h aHi = *(const v8h*)&sA [aBase  + kk + 16];
        v8h bLo = *(const v8h*)&sBt[b0Base + kk];
        v8h bHi = *(const v8h*)&sBt[b0Base + kk + 16];
        v8h cLo = *(const v8h*)&sBt[b1Base + kk];
        v8h cHi = *(const v8h*)&sBt[b1Base + kk + 16];
        v16h a  = __builtin_shufflevector(aLo, aHi, 0,1,2,3,4,5,6,7,8,9,10,11,12,13,14,15);
        v16h b0 = __builtin_shufflevector(bLo, bHi, 0,1,2,3,4,5,6,7,8,9,10,11,12,13,14,15);
        v16h b1 = __builtin_shufflevector(cLo, cHi, 0,1,2,3,4,5,6,7,8,9,10,11,12,13,14,15);
        acc0 = __builtin_amdgcn_wmma_f32_16x16x32_f16(false, a, false, b0,
                                                      (short)0, acc0, false, false);
        acc1 = __builtin_amdgcn_wmma_f32_16x16x32_f16(false, a, false, b1,
                                                      (short)0, acc1, false, false);
    }

    #pragma unroll
    for (int r = 0; r < 8; ++r){
        int mrow = blockRow + wr*16 + r + 8*hi;   // C/D: M = r + 8*(lane>=16), N = lane&15
        if (mrow >= M) continue;
        #pragma unroll
        for (int t2 = 0; t2 < 2; ++t2){
            int col = wc*32 + t2*16 + l15;
            float v = (t2 == 0) ? acc0[r] : acc1[r];
            if (EPI >= 1) v += bias[col];
            if (EPI == 3) v = v * gamma[col] + beta[col];
            if (EPI >= 2) v = silu_f(v);
            out[(long)mrow*128 + col] = v;
        }
    }
}

// xv[n,h,c] = t[n,h] * pos[n,c]
__global__ void xv_init_kernel(const float* __restrict__ t, const float* __restrict__ pos,
                               float* __restrict__ xv, int n_total)
{
    int i = blockIdx.x*blockDim.x + threadIdx.x;
    if (i >= n_total) return;
    int n = i >> 7;
    float v = t[i];
    xv[(long)i*3+0] = v*pos[n*3+0];
    xv[(long)i*3+1] = v*pos[n*3+1];
    xv[(long)i*3+2] = v*pos[n*3+2];
}

// m = silu(As[src] + Bs[dst] + Em + cbm)
__global__ void message_kernel(const float* __restrict__ As, const float* __restrict__ Bs,
                               const float* __restrict__ Em, const float* __restrict__ cbm,
                               const int* __restrict__ ei, float* __restrict__ m, int EH)
{
    int i = blockIdx.x*blockDim.x + threadIdx.x;
    if (i >= EH) return;
    int e = i >> 7, h = i & 127;
    int s = ei[e], d = ei[N_EDGES + e];
    float v = As[(long)s*128+h] + Bs[(long)d*128+h] + Em[i] + cbm[h];
    m[i] = silu_f(v);
}

// sm[dst] += m ; sv[dst,h,:] += xv[src,h,:]*g1 + g2*dp
__global__ void scatter_kernel(const float* __restrict__ m, const float* __restrict__ g1,
                               const float* __restrict__ g2, const float* __restrict__ xv,
                               const float* __restrict__ pos, const int* __restrict__ ei,
                               float* __restrict__ sm, float* __restrict__ sv, int EH)
{
    int i = blockIdx.x*blockDim.x + threadIdx.x;
    if (i >= EH) return;
    int e = i >> 7, h = i & 127;
    int s = ei[e], d = ei[N_EDGES + e];
    atomicAdd(&sm[(long)d*128+h], m[i]);
    float a = g1[i], b = g2[i];
    float dp0 = pos[d*3+0]-pos[s*3+0];
    float dp1 = pos[d*3+1]-pos[s*3+1];
    float dp2 = pos[d*3+2]-pos[s*3+2];
    const float* xp = &xv[(long)(s*128+h)*3];
    atomicAdd(&sv[(long)(d*128+h)*3+0], xp[0]*a + b*dp0);
    atomicAdd(&sv[(long)(d*128+h)*3+1], xp[1]*a + b*dp1);
    atomicAdd(&sv[(long)(d*128+h)*3+2], xp[2]*a + b*dp2);
}

// LayerNorm + equivariant norm + gating + residual. Block per node, 128 threads.
__global__ __launch_bounds__(128)
void finish_kernel(const float* __restrict__ Xc1, const float* __restrict__ Xc2,
                   const float* __restrict__ cbs, const float* __restrict__ xs_res,
                   const float* __restrict__ xv_old, const float* __restrict__ sv,
                   const float* __restrict__ lng, const float* __restrict__ lnb,
                   const float* __restrict__ vsg,
                   float* __restrict__ xs_new, float* __restrict__ xv_new, int first)
{
    int n = blockIdx.x, h = threadIdx.x;
    __shared__ float red[128];
    long base = (long)n*128 + h;
    float xc = Xc1[base] + Xc2[base] + cbs[h];
    float o0 = xv_old[base*3+0], o1 = xv_old[base*3+1], o2 = xv_old[base*3+2];
    float v0 = o0 + sv[base*3+0];
    float v1 = o1 + sv[base*3+1];
    float v2 = o2 + sv[base*3+2];

    red[h] = xc; __syncthreads();
    for (int s=64; s>0; s>>=1){ if (h<s) red[h]+=red[h+s]; __syncthreads(); }
    float mu = red[0] * (1.0f/128.0f); __syncthreads();

    float dd = xc - mu;
    red[h] = dd*dd; __syncthreads();
    for (int s=64; s>0; s>>=1){ if (h<s) red[h]+=red[h+s]; __syncthreads(); }
    float var = red[0] * (1.0f/128.0f); __syncthreads();

    red[h] = v0*v0 + v1*v1 + v2*v2; __syncthreads();
    for (int s=64; s>0; s>>=1){ if (h<s) red[h]+=red[h+s]; __syncthreads(); }
    float nrm = sqrtf(red[0] * (1.0f/128.0f) + EPSV); __syncthreads();

    float xs_n = dd * rsqrtf(var + EPSV) * lng[h] + lnb[h];
    float sg = sigm_f(xs_n);
    float gsc = vsg[h] * sg * rcp_f(nrm);
    float resS = first ? 0.f : xs_res[base];
    xs_new[base] = xs_n*sg + resS;
    xv_new[base*3+0] = v0*gsc + (first ? 0.f : o0);
    xv_new[base*3+1] = v1*gsc + (first ? 0.f : o1);
    xv_new[base*3+2] = v2*gsc + (first ? 0.f : o2);
}

// pvn[n,p] = || sum_h xv[n,h,:] * pvW[h,p] ||  (with 1e-9)
__global__ void pv_norm_kernel(const float* __restrict__ xv, const float* __restrict__ pvW,
                               float* __restrict__ pvn, int total)
{
    int i = blockIdx.x*blockDim.x + threadIdx.x;
    if (i >= total) return;
    int n = i >> 4, p = i & 15;
    float a0=0.f, a1=0.f, a2=0.f;
    const float* xp = &xv[(long)n*128*3];
    for (int hh=0; hh<128; ++hh){
        float w = pvW[hh*16 + p];
        a0 += xp[hh*3+0]*w; a1 += xp[hh*3+1]*w; a2 += xp[hh*3+2]*w;
    }
    pvn[i] = sqrtf(a0*a0 + a1*a1 + a2*a2 + 1e-9f);
}

// u[n,p,:] = squash(pc_lin[n,p,:]) + pvn[n,:]
__global__ void squash_u_kernel(const float* __restrict__ pc_lin, const float* __restrict__ pvn,
                                float* __restrict__ u, int total)
{
    int i = blockIdx.x*blockDim.x + threadIdx.x;
    if (i >= total) return;           // total = N*PC
    int n = i >> 3, p = i & 7;
    float s[16]; float n2 = 0.f;
    for (int d=0; d<16; ++d){ s[d] = pc_lin[(long)n*128 + p*16 + d]; n2 += s[d]*s[d]; }
    float f = (n2 * rcp_f(1.f+n2)) * rsqrtf(n2 + 1e-9f);
    for (int d=0; d<16; ++d) u[(long)n*128 + p*16 + d] = s[d]*f + pvn[n*16 + d];
}

// pooled[batch[n],p,c,e] += sum_d u[n,p,d]*sW[p,d,c,e]; block per node, 256 threads = (c,e)
__global__ __launch_bounds__(256)
void pooled_acc_kernel(const float* __restrict__ u, const float* __restrict__ sW,
                       const int* __restrict__ batch, float* __restrict__ pooled)
{
    int n = blockIdx.x, t = threadIdx.x;
    __shared__ float su[128];
    if (t < 128) su[t] = u[(long)n*128 + t];
    __syncthreads();
    int b = batch[n];
    int c = t >> 5, e = t & 31;
    for (int p=0; p<PC; ++p){
        float acc = 0.f;
        for (int d=0; d<PD; ++d) acc += su[p*16+d] * sW[((p*16+d)*8 + c)*32 + e];
        atomicAdd(&pooled[(((long)b*PC + p)*SC + c)*SD + e], acc);
    }
}

// Dynamic routing (2 iters) + attention pooling + 3-layer MLP head. Block per graph.
__global__ __launch_bounds__(256)
void routing_kernel(const float* __restrict__ pooled,
                    const float* __restrict__ attW, const float* __restrict__ attB,
                    const float* __restrict__ pW1, const float* __restrict__ pb1,
                    const float* __restrict__ pg1, const float* __restrict__ pbe1,
                    const float* __restrict__ pW2, const float* __restrict__ pb2,
                    const float* __restrict__ pg2, const float* __restrict__ pbe2,
                    const float* __restrict__ pW3, const float* __restrict__ pb3,
                    float* __restrict__ out)
{
    int b = blockIdx.x, t = threadIdx.x;
    __shared__ float P[PC][SC][SD];
    __shared__ float bb[PC][SC];
    __shared__ float cc[PC][SC];
    __shared__ float vt[SC][SD];
    __shared__ float vv[SC][SD];
    __shared__ float sfac[SC];
    __shared__ float aw[SC];
    __shared__ float wcv[SD];
    __shared__ float h1[64];
    __shared__ float h2[32];

    for (int idx=t; idx<PC*SC*SD; idx+=256)
        ((float*)P)[idx] = pooled[(long)b*PC*SC*SD + idx];
    if (t < PC*SC) ((float*)bb)[t] = 0.f;
    __syncthreads();

    int c = t >> 5, e = t & 31;
    for (int it=0; it<2; ++it){
        if (t < PC){
            int p = t; float mx = bb[p][0];
            for (int j=1;j<SC;++j) mx = fmaxf(mx, bb[p][j]);
            float sum = 0.f;
            for (int j=0;j<SC;++j){ float ex = __expf(bb[p][j]-mx); cc[p][j]=ex; sum+=ex; }
            float rs = rcp_f(sum);
            for (int j=0;j<SC;++j) cc[p][j] *= rs;
        }
        __syncthreads();
        float acc = 0.f;
        for (int p=0;p<PC;++p) acc += cc[p][c]*P[p][c][e];
        vt[c][e] = acc;
        __syncthreads();
        if (t < SC){
            float n2 = 0.f;
            for (int j=0;j<SD;++j){ float x = vt[t][j]; n2 += x*x; }
            sfac[t] = (n2 * rcp_f(1.f+n2)) * rsqrtf(n2 + 1e-9f);
        }
        __syncthreads();
        vv[c][e] = vt[c][e]*sfac[c];
        __syncthreads();
        if (it==0 && t < PC*SC){
            int p = t >> 3, j = t & 7;
            float s = 0.f;
            for (int k=0;k<SD;++k) s += P[p][j][k]*vv[j][k];
            bb[p][j] += s;
        }
        __syncthreads();
    }
    if (t < SC){
        float lg = attB[0];
        for (int d=0; d<SD/2; ++d) lg += vv[t][d]*attW[d];
        sfac[t] = lg;
    }
    __syncthreads();
    if (t == 0){
        float mx = sfac[0]; for (int j=1;j<SC;++j) mx = fmaxf(mx, sfac[j]);
        float sum = 0.f;
        for (int j=0;j<SC;++j){ aw[j] = __expf(sfac[j]-mx); sum += aw[j]; }
        float rs = rcp_f(sum);
        for (int j=0;j<SC;++j) aw[j] *= rs;
    }
    __syncthreads();
    if (t < SD){
        float s = 0.f;
        for (int j=0;j<SC;++j) s += aw[j]*vv[j][t];
        wcv[t] = s;
    }
    __syncthreads();
    if (t < 64){
        float s = pb1[t];
        for (int k=0;k<SD;++k) s += wcv[k]*pW1[k*64+t];
        h1[t] = silu_f(s*pg1[t] + pbe1[t]);
    }
    __syncthreads();
    if (t < 32){
        float s = pb2[t];
        for (int k=0;k<64;++k) s += h1[k]*pW2[k*32+t];
        h2[t] = silu_f(s*pg2[t] + pbe2[t]);
    }
    __syncthreads();
    if (t == 0){
        float s = pb3[0];
        for (int k=0;k<32;++k) s += h2[k]*pW3[k];
        out[b] = s;
    }
}

// ---------------------------------------------------------------------------
extern "C" void kernel_launch(void* const* d_in, const int* in_sizes, int n_in,
                              void* d_out, int out_size, void* d_ws, size_t ws_size,
                              hipStream_t stream)
{
    (void)in_sizes; (void)n_in; (void)out_size; (void)ws_size;
    const float* x         = (const float*)d_in[0];
    const float* edge_attr = (const float*)d_in[1];
    const float* pos       = (const float*)d_in[2];
    const int*   ei        = (const int*)d_in[3];
    const int*   batch     = (const int*)d_in[4];
    const float* neW1=(const float*)d_in[5],  *neb1=(const float*)d_in[6];
    const float* neg1=(const float*)d_in[7],  *nebe1=(const float*)d_in[8];
    const float* neW2=(const float*)d_in[9],  *neb2=(const float*)d_in[10];
    const float* neg2=(const float*)d_in[11], *nebe2=(const float*)d_in[12];
    const float* eeW1=(const float*)d_in[13], *eeb1=(const float*)d_in[14];
    const float* eeW2=(const float*)d_in[15], *eeb2=(const float*)d_in[16];
    const float* ivW =(const float*)d_in[17], *ivb =(const float*)d_in[18];
    const float* cWma=(const float*)d_in[19], *cWmb=(const float*)d_in[20];
    const float* cWme=(const float*)d_in[21], *cbm =(const float*)d_in[22];
    const float* cWsx=(const float*)d_in[23], *cWsa=(const float*)d_in[24];
    const float* cbs =(const float*)d_in[25];
    const float* cWv1=(const float*)d_in[26], *cbv1=(const float*)d_in[27];
    const float* cWv2=(const float*)d_in[28], *cbv2=(const float*)d_in[29];
    const float* lng =(const float*)d_in[30], *lnb =(const float*)d_in[31];
    const float* vsg =(const float*)d_in[32];
    const float* pcW =(const float*)d_in[33], *pcb =(const float*)d_in[34];
    const float* pvW =(const float*)d_in[35], *sW  =(const float*)d_in[36];
    const float* aW  =(const float*)d_in[37], *ab  =(const float*)d_in[38];
    const float* pW1 =(const float*)d_in[39], *pb1 =(const float*)d_in[40];
    const float* pg1 =(const float*)d_in[41], *pbe1=(const float*)d_in[42];
    const float* pW2 =(const float*)d_in[43], *pb2 =(const float*)d_in[44];
    const float* pg2 =(const float*)d_in[45], *pbe2=(const float*)d_in[46];
    const float* pW3 =(const float*)d_in[47], *pb3 =(const float*)d_in[48];
    float* out = (float*)d_out;

    // workspace carve-up (f32)
    size_t off = 0;
    auto wsf = [&](size_t n){ float* p = (float*)((char*)d_ws + off); off += n*sizeof(float); return p; };
    float* ea  = wsf((size_t)N_EDGES*H);     // edge embedding (persistent)
    float* ebA = wsf((size_t)N_EDGES*H);     // m (also edge-embed intermediate)
    float* ebB = wsf((size_t)N_EDGES*H);     // Em -> g1
    float* ebC = wsf((size_t)N_EDGES*H);     // g2
    float* xsA = wsf((size_t)N_NODES*H);
    float* xsB = wsf((size_t)N_NODES*H);
    float* xvA = wsf((size_t)N_NODES*H*3);
    float* xvB = wsf((size_t)N_NODES*H*3);
    float* sv  = wsf((size_t)N_NODES*H*3);
    float* nb1 = wsf((size_t)N_NODES*H);     // As / sm / pc_lin
    float* nb2 = wsf((size_t)N_NODES*H);     // Bs / Xc1 / u
    float* nb3 = wsf((size_t)N_NODES*H);     // Xc2
    float* pvn = wsf((size_t)N_NODES*PD);
    float* pooled = wsf((size_t)BGR*PC*SC*SD);

    dim3 blk(256);
    const int gN  = (N_NODES + 31) / 32;
    const int gE  = (N_EDGES + 31) / 32;
    const int EH  = N_EDGES * H;
    const int NH  = N_NODES * H;
    const int gEH = (EH + 255) / 256;
    const int gNH = (NH + 255) / 256;

    // node embedding: silu(bn(x@neW1)) -> silu(bn(.@neW2))
    gemm128_wmma<3><<<gN, blk, 0, stream>>>(x,   FN, FN, neW1, neb1, neg1, nebe1, nb1, N_NODES);
    gemm128_wmma<3><<<gN, blk, 0, stream>>>(nb1, H,  H,  neW2, neb2, neg2, nebe2, xsA, N_NODES);
    // edge embedding: silu(.@eeW1+b) -> silu(.@eeW2+b)
    gemm128_wmma<2><<<gE, blk, 0, stream>>>(edge_attr, FE, FE, eeW1, eeb1, nullptr, nullptr, ebA, N_EDGES);
    gemm128_wmma<2><<<gE, blk, 0, stream>>>(ebA,       H,  H,  eeW2, eeb2, nullptr, nullptr, ea,  N_EDGES);
    // xv = (xs@ivW+ivb) outer pos
    gemm128_wmma<1><<<gN, blk, 0, stream>>>(xsA, H, H, ivW, ivb, nullptr, nullptr, nb1, N_NODES);
    xv_init_kernel<<<gNH, blk, 0, stream>>>(nb1, pos, xvA, NH);

    float* xs_cur = xsA; float* xs_nxt = xsB;
    float* xv_cur = xvA; float* xv_nxt = xvB;
    for (int i = 0; i < LYR; ++i){
        const float* Wma = cWma + (size_t)i*H*H;
        const float* Wmb = cWmb + (size_t)i*H*H;
        const float* Wme = cWme + (size_t)i*H*H;
        const float* Wsx = cWsx + (size_t)i*H*H;
        const float* Wsa = cWsa + (size_t)i*H*H;
        const float* Wv1 = cWv1 + (size_t)i*H*H;
        const float* Wv2 = cWv2 + (size_t)i*H*H;

        // node-side projections (avoids per-edge GEMM for xs[src]/xs[dst])
        gemm128_wmma<0><<<gN, blk, 0, stream>>>(xs_cur, H, H, Wma, nullptr, nullptr, nullptr, nb1, N_NODES);
        gemm128_wmma<0><<<gN, blk, 0, stream>>>(xs_cur, H, H, Wmb, nullptr, nullptr, nullptr, nb2, N_NODES);
        gemm128_wmma<0><<<gE, blk, 0, stream>>>(ea,     H, H, Wme, nullptr, nullptr, nullptr, ebB, N_EDGES);
        message_kernel<<<gEH, blk, 0, stream>>>(nb1, nb2, ebB, cbm + i*H, ei, ebA, EH);
        // vector-message gates
        gemm128_wmma<1><<<gE, blk, 0, stream>>>(ebA, H, H, Wv1, cbv1 + i*H, nullptr, nullptr, ebB, N_EDGES);
        gemm128_wmma<1><<<gE, blk, 0, stream>>>(ebA, H, H, Wv2, cbv2 + i*H, nullptr, nullptr, ebC, N_EDGES);
        // scatter-add message aggregation
        hipMemsetAsync(nb1, 0, (size_t)NH*sizeof(float), stream);
        hipMemsetAsync(sv,  0, (size_t)NH*3*sizeof(float), stream);
        scatter_kernel<<<gEH, blk, 0, stream>>>(ebA, ebB, ebC, xv_cur, pos, ei, nb1, sv, EH);
        // scalar update projections
        gemm128_wmma<0><<<gN, blk, 0, stream>>>(xs_cur, H, H, Wsx, nullptr, nullptr, nullptr, nb2, N_NODES);
        gemm128_wmma<0><<<gN, blk, 0, stream>>>(nb1,    H, H, Wsa, nullptr, nullptr, nullptr, nb3, N_NODES);
        // norms + gating + residual
        finish_kernel<<<N_NODES, 128, 0, stream>>>(nb2, nb3, cbs + i*H, xs_cur, xv_cur, sv,
                                                   lng + i*H, lnb + i*H, vsg + i*H,
                                                   xs_nxt, xv_nxt, (i == 0) ? 1 : 0);
        float* t1 = xs_cur; xs_cur = xs_nxt; xs_nxt = t1;
        float* t2 = xv_cur; xv_cur = xv_nxt; xv_nxt = t2;
    }

    // primary capsules
    gemm128_wmma<1><<<gN, blk, 0, stream>>>(xs_cur, H, H, pcW, pcb, nullptr, nullptr, nb1, N_NODES);
    pv_norm_kernel<<<(N_NODES*PD + 255)/256, blk, 0, stream>>>(xv_cur, pvW, pvn, N_NODES*PD);
    squash_u_kernel<<<(N_NODES*PC + 255)/256, blk, 0, stream>>>(nb1, pvn, nb2, N_NODES*PC);

    // per-graph pooled capsule tensor
    hipMemsetAsync(pooled, 0, (size_t)BGR*PC*SC*SD*sizeof(float), stream);
    pooled_acc_kernel<<<N_NODES, 256, 0, stream>>>(nb2, sW, batch, pooled);

    // routing + attention + head MLP
    routing_kernel<<<BGR, 256, 0, stream>>>(pooled, aW, ab,
                                            pW1, pb1, pg1, pbe1,
                                            pW2, pb2, pg2, pbe2,
                                            pW3, pb3, out);
}
